// ResBlock_1_77137612636474
// MI455X (gfx1250) — compile-verified, matrix operand order checked by reading, not run on hardware
//
#include <hip/hip_runtime.h>
#include <hip/hip_bf16.h>

// ============================================================================
// Binarized residual block on gfx1250 (MI455X).
//   y1 = BN1( bconv(sign(x),  sign(w1)) )
//   y2 = BN2( bconv(sign(y1), sign(w2)) )
//   out = x + y2
// Convs are exact integer GEMMs done with V_WMMA_I32_16X16X64_IU8.
// K dimension = 288 (32 ch x 9 taps), split into 5 chunks of 64 (last padded).
// Intermediates stored int16 NHWC (|val| <= 288). BN stats: shuffle-reduce in
// registers -> LDS u64 atomics (lanes 0/16 only) -> global u64 atomics; exact
// and deterministic across graph replays.
// Each wave owns TWO 16-pixel tiles -> 4 independent WMMA accumulator chains
// so the scheduler can fill the IU8 WMMA hazard slots (8 co-exec NOPs) with
// useful work instead of v_nop.
// ============================================================================

typedef int v8i __attribute__((ext_vector_type(8)));

#define N_    16
#define C_    32
#define H_    224
#define W_    224
#define KTAPS 9
#define KTOT  (C_ * KTAPS)      // 288
#define NCHUNK 5                // ceil(288/64)
#define PIX   (N_ * H_ * W_)    // 802816

// LDS staging tile: 3 halo rows x 226 cols x channels padded 32->36 bytes
#define LCOLS 226
#define LCH   36

// ---- workspace layout (bytes) ----
#define WS_A1   0                                   // 5*2*32*8 dwords = 10240 B
#define WS_A2   10240
#define WS_SUM1 20480                               // 32 x u64
#define WS_SQ1  (20480 + 256)
#define WS_SUM2 (20480 + 512)
#define WS_SQ2  (20480 + 768)
#define WS_BN1S 21504                               // 32 x f32 each
#define WS_BN1B (21504 + 128)
#define WS_BN2S (21504 + 256)
#define WS_BN2B (21504 + 384)
#define WS_C1   24576                               // int16 NHWC, 51.4 MB
#define WS_C2   (24576 + (size_t)PIX * C_ * 2)

// A-fragment K mapping for 8-bit 16x64 A tiles (CDNA5 ISA 7.12.2):
//   K(v,lane,byte) = 16*(v>>1) + 4*(v&1) + (lane>=16 ? 8 : 0) + byte
// B (64x16, 8-bit) assumed mirror layout with N = lane&15.

__global__ void init_stats_kernel(unsigned long long* p) {
    int t = blockIdx.x * blockDim.x + threadIdx.x;
    if (t < 128) p[t] = 0ull;   // sum1, sq1, sum2, sq2 contiguous
}

// Pack sign(w) into WMMA A-fragment register layout for both convs.
__global__ void pack_weights_kernel(const float* __restrict__ w1,
                                    const float* __restrict__ w2,
                                    int* __restrict__ A1, int* __restrict__ A2) {
    int t = blockIdx.x * blockDim.x + threadIdx.x;
    const int total = 2 * NCHUNK * 2 * 32 * 8;      // = 5120 dwords
    for (int id = t; id < total; id += gridDim.x * blockDim.x) {
        int v  = id & 7;
        int l  = (id >> 3) & 31;
        int mh = (id >> 8) & 1;
        int kc = (id >> 9) % NCHUNK;
        int cv = (id >> 9) / NCHUNK;
        const float* wsrc = cv ? w2 : w1;
        int*         dst  = cv ? A2 : A1;
        int base = ((v >> 1) << 4) + ((v & 1) << 2) + ((l & 16) ? 8 : 0);
        int M = mh * 16 + (l & 15);
        unsigned int dw = 0;
        #pragma unroll
        for (int b = 0; b < 4; ++b) {
            int Kg = kc * 64 + base + b;
            int s8 = 0;
            if (Kg < KTOT) {
                int tap = Kg >> 5, ci = Kg & 31;        // K = tap*32 + ci
                float wv = wsrc[(M * C_ + ci) * KTAPS + tap];
                s8 = (wv > 0.f) - (wv < 0.f);           // sign(0)=0, matches jnp
            }
            dw |= ((unsigned int)(unsigned char)(signed char)s8) << (8 * b);
        }
        dst[((kc * 2 + mh) * 32 + l) * 8 + v] = (int)dw;
    }
}

// Build one B-fragment dword set for the pixel column `wpix` and chunk kc.
__device__ __forceinline__ v8i load_bfrag(const signed char* s_act, int lane,
                                          int wpix, int kc) {
    v8i bv;
    #pragma unroll
    for (int v = 0; v < 8; ++v) {
        int base = ((v >> 1) << 4) + ((v & 1) << 2) + ((lane & 16) ? 8 : 0);
        int Kg = kc * 64 + base;                 // 4 consecutive K per dword
        int word = 0;
        if (Kg < KTOT) {                         // chunk 4 is half-padded
            int tap = Kg >> 5, ci = Kg & 31;     // same tap, ci..ci+3
            int dy = tap / 3, dx = tap % 3;
            word = *(const int*)(s_act + (dy * LCOLS + wpix + dx) * LCH + ci);
        }
        bv[v] = word;
    }
    return bv;
}

// One block = one (n, h) output row of 224 pixels; 7 waves x 2 tiles x 16 px.
// mode 0: stage sign(x) from fp32 NCHW.  mode 1: stage sign(bn(q)) int16 NHWC.
__global__ __launch_bounds__(224) void bconv_kernel(
    const float* __restrict__ xin, const short* __restrict__ qin, int mode,
    const float* __restrict__ bnScale, const float* __restrict__ bnBias,
    const int* __restrict__ wsA, short* __restrict__ outq,
    unsigned long long* __restrict__ gSum, unsigned long long* __restrict__ gSq)
{
    __shared__ signed char s_act[3 * LCOLS * LCH];   // 24408 B
    __shared__ unsigned long long s_sum[C_], s_sq[C_];

    const int h = blockIdx.x, n = blockIdx.y;
    const int tid = threadIdx.x;
    if (tid < C_) { s_sum[tid] = 0ull; s_sq[tid] = 0ull; }

    // ---- stage binarized activations (3 rows, halo cols, 32 ch) into LDS ----
    if (mode == 0) {
        for (int id = tid; id < 3 * LCOLS * C_; id += 224) {
            int c = id / (3 * LCOLS);
            int rem = id % (3 * LCOLS);
            int r = rem / LCOLS, col = rem % LCOLS;
            int hg = h - 1 + r, wg = col - 1;
            signed char s = 0;
            if (hg >= 0 && hg < H_ && wg >= 0 && wg < W_) {
                float v = xin[(((size_t)n * C_ + c) * H_ + hg) * W_ + wg];
                s = (signed char)((v > 0.f) - (v < 0.f));
            }
            s_act[(r * LCOLS + col) * LCH + c] = s;
        }
    } else {
        for (int id = tid; id < 3 * LCOLS * C_; id += 224) {
            int c = id & 31;
            int col = (id >> 5) % LCOLS;
            int r = (id >> 5) / LCOLS;
            int hg = h - 1 + r, wg = col - 1;
            signed char s = 0;
            if (hg >= 0 && hg < H_ && wg >= 0 && wg < W_) {
                float v = bnScale[c] *
                          (float)qin[(((size_t)n * H_ + hg) * W_ + wg) * C_ + c] +
                          bnBias[c];
                s = (signed char)((v > 0.f) - (v < 0.f));
            }
            s_act[(r * LCOLS + col) * LCH + c] = s;
        }
    }
    __syncthreads();

    // ---- WMMA: 2 tiles x 16 pixels x 32 out-channels per wave -------------
    const int lane  = tid & 31;
    const int wv    = tid >> 5;
    const int wpixA = wv * 32 + (lane & 15);         // tile A pixel column
    const int wpixB = wpixA + 16;                    // tile B pixel column
    v8i accA0 = {0,0,0,0,0,0,0,0}, accA1 = accA0;
    v8i accB0 = accA0,             accB1 = accA0;

    #pragma unroll
    for (int kc = 0; kc < NCHUNK; ++kc) {
        v8i a0 = *(const v8i*)(wsA + ((kc * 2 + 0) * 32 + lane) * 8);
        v8i a1 = *(const v8i*)(wsA + ((kc * 2 + 1) * 32 + lane) * 8);
        v8i bA = load_bfrag(s_act, lane, wpixA, kc);
        v8i bB = load_bfrag(s_act, lane, wpixB, kc);
        // 4 independent chains -> scheduler fills IU8 hazard slots
        accA0 = __builtin_amdgcn_wmma_i32_16x16x64_iu8(true, a0, true, bA, accA0,
                                                       false, false);
        accB0 = __builtin_amdgcn_wmma_i32_16x16x64_iu8(true, a0, true, bB, accB0,
                                                       false, false);
        accA1 = __builtin_amdgcn_wmma_i32_16x16x64_iu8(true, a1, true, bA, accA1,
                                                       false, false);
        accB1 = __builtin_amdgcn_wmma_i32_16x16x64_iu8(true, a1, true, bB, accB1,
                                                       false, false);
    }

    // ---- store int16 NHWC ---------------------------------------------------
    const size_t pbaseA = (((size_t)n * H_ + h) * W_ + wpixA) * C_;
    const size_t pbaseB = (((size_t)n * H_ + h) * W_ + wpixB) * C_;
    const int choff = (lane & 16) ? 8 : 0;           // D layout: lanes16-31 = M+8
    #pragma unroll
    for (int v = 0; v < 8; ++v) {
        outq[pbaseA + (v + choff)]      = (short)accA0[v];
        outq[pbaseA + (16 + v + choff)] = (short)accA1[v];
        outq[pbaseB + (v + choff)]      = (short)accB0[v];
        outq[pbaseB + (16 + v + choff)] = (short)accB1[v];
    }

    // ---- stats: butterfly-reduce over each 16-lane half, then LDS atomics --
    // Channel (v+choff) lives on one 16-lane half; masks 1,2,4,8 stay inside.
    #pragma unroll
    for (int v = 0; v < 8; ++v) {
        int s0 = accA0[v] + accB0[v];
        int q0 = accA0[v] * accA0[v] + accB0[v] * accB0[v];
        int s1 = accA1[v] + accB1[v];
        int q1 = accA1[v] * accA1[v] + accB1[v] * accB1[v];
        #pragma unroll
        for (int m = 1; m <= 8; m <<= 1) {
            s0 += __shfl_xor(s0, m, 32);
            q0 += __shfl_xor(q0, m, 32);
            s1 += __shfl_xor(s1, m, 32);
            q1 += __shfl_xor(q1, m, 32);
        }
        if ((lane & 15) == 0) {
            atomicAdd(&s_sum[v + choff],      (unsigned long long)(long long)s0);
            atomicAdd(&s_sq [v + choff],      (unsigned long long)(long long)q0);
            atomicAdd(&s_sum[16 + v + choff], (unsigned long long)(long long)s1);
            atomicAdd(&s_sq [16 + v + choff], (unsigned long long)(long long)q1);
        }
    }
    __syncthreads();
    if (tid < C_) {
        atomicAdd(&gSum[tid], s_sum[tid]);
        atomicAdd(&gSq[tid],  s_sq[tid]);
    }
}

__global__ void bn_finalize_kernel(const unsigned long long* __restrict__ gSum,
                                   const unsigned long long* __restrict__ gSq,
                                   const float* __restrict__ gamma,
                                   const float* __restrict__ beta,
                                   float* __restrict__ scale,
                                   float* __restrict__ bias) {
    int c = threadIdx.x;
    if (c < C_) {
        double cnt  = (double)PIX;
        double mean = (double)(long long)gSum[c] / cnt;
        double var  = (double)(long long)gSq[c] / cnt - mean * mean;
        float inv = rsqrtf((float)var + 1e-5f);
        float sc  = gamma[c] * inv;
        scale[c] = sc;
        bias[c]  = beta[c] - (float)mean * sc;
    }
}

__global__ void residual_out_kernel(const float* __restrict__ x,
                                    const short* __restrict__ q2,
                                    const float* __restrict__ scale,
                                    const float* __restrict__ bias,
                                    float* __restrict__ out) {
    size_t idx = (size_t)blockIdx.x * blockDim.x + threadIdx.x;
    const size_t total = (size_t)N_ * C_ * H_ * W_;
    if (idx >= total) return;
    int w = (int)(idx % W_);
    int h = (int)((idx / W_) % H_);
    int c = (int)((idx / ((size_t)W_ * H_)) % C_);
    int n = (int)(idx / ((size_t)W_ * H_ * C_));
    float v = (float)q2[(((size_t)n * H_ + h) * W_ + w) * C_ + c];
    out[idx] = x[idx] + scale[c] * v + bias[c];
}

extern "C" void kernel_launch(void* const* d_in, const int* in_sizes, int n_in,
                              void* d_out, int out_size, void* d_ws, size_t ws_size,
                              hipStream_t stream) {
    const float* x  = (const float*)d_in[0];
    const float* w1 = (const float*)d_in[1];
    const float* g1 = (const float*)d_in[2];
    const float* b1 = (const float*)d_in[3];
    const float* w2 = (const float*)d_in[4];
    const float* g2 = (const float*)d_in[5];
    const float* b2 = (const float*)d_in[6];

    char* ws = (char*)d_ws;
    int* A1 = (int*)(ws + WS_A1);
    int* A2 = (int*)(ws + WS_A2);
    unsigned long long* sum1 = (unsigned long long*)(ws + WS_SUM1);
    unsigned long long* sq1  = (unsigned long long*)(ws + WS_SQ1);
    unsigned long long* sum2 = (unsigned long long*)(ws + WS_SUM2);
    unsigned long long* sq2  = (unsigned long long*)(ws + WS_SQ2);
    float* sc1 = (float*)(ws + WS_BN1S);
    float* bi1 = (float*)(ws + WS_BN1B);
    float* sc2 = (float*)(ws + WS_BN2S);
    float* bi2 = (float*)(ws + WS_BN2B);
    short* c1 = (short*)(ws + WS_C1);
    short* c2 = (short*)(ws + WS_C2);
    float* out = (float*)d_out;

    init_stats_kernel<<<1, 128, 0, stream>>>(sum1);          // zeros all 4 stat arrays
    pack_weights_kernel<<<20, 256, 0, stream>>>(w1, w2, A1, A2);

    dim3 grid(H_, N_);
    bconv_kernel<<<grid, 224, 0, stream>>>(x, nullptr, 0, nullptr, nullptr,
                                           A1, c1, sum1, sq1);
    bn_finalize_kernel<<<1, 32, 0, stream>>>(sum1, sq1, g1, b1, sc1, bi1);

    bconv_kernel<<<grid, 224, 0, stream>>>(nullptr, c1, 1, sc1, bi1,
                                           A2, c2, sum2, sq2);
    bn_finalize_kernel<<<1, 32, 0, stream>>>(sum2, sq2, g2, b2, sc2, bi2);

    const size_t total = (size_t)N_ * C_ * H_ * W_;
    residual_out_kernel<<<(unsigned)((total + 255) / 256), 256, 0, stream>>>(
        x, c2, sc2, bi2, out);
}
